// PolicyValueDWPixelPlayerAttnNoFFNNet_17703855194289
// MI455X (gfx1250) — matrix operand order repaired and checked by reading, not compiled, not run on hardware
//
#include <hip/hip_runtime.h>

// ---------------------------------------------------------------------------
// Problem constants (from reference): B=16, H=W=32 (HW=1024), M=8 players,
// hid C=64, HEADS=4 (hd=16), L=6 layers, board channels 149, stem in = 35.
// Token rows R = B*HW*M = 131072, pixel-major row index = (b*1024+hw)*8 + m.
// ---------------------------------------------------------------------------
#define RTOK   131072
#define CH     64
#define LAYERS 6

typedef __attribute__((ext_vector_type(16))) __bf16        v16bf;
typedef __attribute__((ext_vector_type(8)))  float         v8f;
typedef __attribute__((ext_vector_type(8)))  unsigned int  v8u;

__device__ __forceinline__ unsigned short f2bf(float f) {
    unsigned int u = __builtin_bit_cast(unsigned int, f);
    unsigned int r = u + 0x7FFFu + ((u >> 16) & 1u);   // round-to-nearest-even
    return (unsigned short)(r >> 16);
}
__device__ __forceinline__ float siluf(float x) { return x / (1.0f + __expf(-x)); }

// --- WMMA fragment loaders --------------------------------------------------
// A matrix 16x32 bf16: lanes 0-15 and 16-31 both hold rows M=0..15;
// per lane 8 dwords: dword d -> K = kblk + (d/4)*16 + (lane>=16 ? 8:0) + 2*(d%4).
__device__ __forceinline__ v16bf load_a_frag(const unsigned short* __restrict__ A,
                                             int row, int kblk, int lane) {
    const unsigned int* Ar =
        (const unsigned int*)(A + row * CH + kblk + ((lane & 16) ? 8 : 0));
    v8u t;
    t[0] = Ar[0]; t[1] = Ar[1]; t[2] = Ar[2];  t[3] = Ar[3];    // K group 0
    t[4] = Ar[8]; t[5] = Ar[9]; t[6] = Ar[10]; t[7] = Ar[11];   // K group 1 (+16)
    return __builtin_bit_cast(v16bf, t);
}
// B matrix 32x16 bf16 (B[k][n] = W[n][k], W row-major 64x64):
// lane holds N = nt*16 + (lane&15); dword d -> K = kblk + (d/4)*16 + (lane>=16 ? 8:0) + 2*(d%4).
__device__ __forceinline__ v16bf load_b_frag(const unsigned short* __restrict__ W,
                                             int nt, int kblk, int lane) {
    const unsigned int* Wr = (const unsigned int*)
        (W + (nt * 16 + (lane & 15)) * CH + kblk + ((lane & 16) ? 8 : 0));
    v8u t;
#pragma unroll
    for (int i = 0; i < 8; ++i) t[i] = Wr[(i < 4) ? i : i + 4];
    return __builtin_bit_cast(v16bf, t);
}

// ---------------------------------------------------------------------------
// present mask: m_each[b] = argmax(board[b,7:14,0,0]) + 2
__global__ void k_present(const float* __restrict__ board, int* __restrict__ present) {
    int b = threadIdx.x;
    if (b >= 16) return;
    float best = board[(b * 149 + 7) * 1024];
    int arg = 0;
#pragma unroll
    for (int j = 1; j < 7; ++j) {
        float v = board[(b * 149 + 7 + j) * 1024];
        if (v > best) { best = v; arg = j; }
    }
    int me = arg + 2;
#pragma unroll
    for (int m = 0; m < 8; ++m) present[b * 8 + m] = (m < me) ? 1 : 0;
}

// Convert all 64x64 weight matrices to bf16. Slot 0 = stem (zero padded 35->64),
// slots 1+l*5+{0..4} = qw,kw,vw,ow,pw for layer l.
__global__ void k_prep_w(const float* __restrict__ stem_w, const float* __restrict__ qw,
                         const float* __restrict__ kw,     const float* __restrict__ vw,
                         const float* __restrict__ ow,     const float* __restrict__ pw,
                         unsigned short* __restrict__ Wbf) {
    int i = blockIdx.x * 256 + threadIdx.x;
    if (i >= 31 * 4096) return;
    int slot = i >> 12, e = i & 4095, o = e >> 6, c = e & 63;
    float val;
    if (slot == 0) {
        val = (c < 35) ? stem_w[o * 35 + c] : 0.0f;
    } else {
        int l = (slot - 1) / 5, which = (slot - 1) % 5;
        const float* base = (which == 0) ? qw : (which == 1) ? kw
                          : (which == 2) ? vw : (which == 3) ? ow : pw;
        val = base[l * 4096 + e];
    }
    Wbf[i] = f2bf(val);
}

// Gather stem input (g 19ch + per-player 16ch, zero padded to 64) as bf16 rows.
__global__ void k_stem_gather(const float* __restrict__ board,
                              unsigned short* __restrict__ Abf) {
    int idx = blockIdx.x * 256 + threadIdx.x;          // RTOK*64 threads
    int c = idx & 63, row = idx >> 6;
    int m = row & 7, bn = row >> 3, hw = bn & 1023, b = bn >> 10;
    float val = 0.0f;
    if (c < 19)      val = board[(b * 149 + c) * 1024 + hw];
    else if (c < 35) val = board[(b * 149 + 19 + m * 16 + (c - 19)) * 1024 + hw];
    Abf[idx] = f2bf(val);
}

// Generic WMMA GEMM: out(R x 64, f32) = A(R x 64, bf16) @ W^T(64 x 64, bf16) + bias
__global__ __launch_bounds__(128)
void k_gemm(const unsigned short* __restrict__ A, const unsigned short* __restrict__ W,
            const float* __restrict__ bias, float* __restrict__ out, int R) {
    int wid = threadIdx.x >> 5, lane = threadIdx.x & 31;
    int r0 = (blockIdx.x * 4 + wid) * 16;
    if (r0 >= R) return;
    __builtin_prefetch(A + (r0 + 64) * CH, 0, 1);
    v16bf a0 = load_a_frag(A, r0 + (lane & 15), 0,  lane);
    v16bf a1 = load_a_frag(A, r0 + (lane & 15), 32, lane);
#pragma unroll
    for (int nt = 0; nt < 4; ++nt) {
        int n = nt * 16 + (lane & 15);
        float bv = bias ? bias[n] : 0.0f;
        v8f c = {bv, bv, bv, bv, bv, bv, bv, bv};
        v16bf b0 = load_b_frag(W, nt, 0,  lane);
        v16bf b1 = load_b_frag(W, nt, 32, lane);
        c = __builtin_amdgcn_wmma_f32_16x16x32_bf16(false, a0, false, b0, (short)0, c, false, false);
        c = __builtin_amdgcn_wmma_f32_16x16x32_bf16(false, a1, false, b1, (short)0, c, false, false);
        int rb = r0 + ((lane & 16) ? 8 : 0);
#pragma unroll
        for (int i = 0; i < 8; ++i) out[(rb + i) * CH + n] = c[i];
    }
}

// Fused per-pixel attention: QKV WMMA -> LDS, masked softmax over M=8 keys,
// O-projection WMMA, bias + residual + LayerNorm epilogue (in-place on tok).
__global__ __launch_bounds__(128)
void k_attn(const unsigned short* __restrict__ Abf, float* __restrict__ tok,
            const unsigned short* __restrict__ Wq, const unsigned short* __restrict__ Wk,
            const unsigned short* __restrict__ Wv, const unsigned short* __restrict__ Wo,
            const float* __restrict__ qb, const float* __restrict__ kbias,
            const float* __restrict__ vb, const float* __restrict__ ob,
            const float* __restrict__ lng, const float* __restrict__ lnb,
            const int* __restrict__ present) {
    __shared__ float sQ[4][16][CH];
    __shared__ float sK[4][16][CH];
    __shared__ float sV[4][16][CH];
    int wid = threadIdx.x >> 5, lane = threadIdx.x & 31;
    int r0 = (blockIdx.x * 4 + wid) * 16;          // 2 pixels x 8 players
    int b  = r0 >> 13;                             // 8192 rows per batch
    v16bf a0 = load_a_frag(Abf, r0 + (lane & 15), 0,  lane);
    v16bf a1 = load_a_frag(Abf, r0 + (lane & 15), 32, lane);

    const unsigned short* Ws[3] = {Wq, Wk, Wv};
    const float*          Bs[3] = {qb, kbias, vb};
    for (int m3 = 0; m3 < 3; ++m3) {
        float (*dst)[CH] = (m3 == 0) ? sQ[wid] : (m3 == 1) ? sK[wid] : sV[wid];
#pragma unroll
        for (int nt = 0; nt < 4; ++nt) {
            int n = nt * 16 + (lane & 15);
            float bv = Bs[m3][n];
            v8f c = {bv, bv, bv, bv, bv, bv, bv, bv};
            v16bf b0 = load_b_frag(Ws[m3], nt, 0,  lane);
            v16bf b1 = load_b_frag(Ws[m3], nt, 32, lane);
            c = __builtin_amdgcn_wmma_f32_16x16x32_bf16(false, a0, false, b0, (short)0, c, false, false);
            c = __builtin_amdgcn_wmma_f32_16x16x32_bf16(false, a1, false, b1, (short)0, c, false, false);
            int rb = (lane & 16) ? 8 : 0;
#pragma unroll
            for (int i = 0; i < 8; ++i) dst[rb + i][n] = c[i];
        }
    }
    __syncthreads();

    // 16 rows x 4 heads = 64 items, 2 per lane.
    float ao[2][16];
#pragma unroll
    for (int it = 0; it < 2; ++it) {
        int rh = lane + it * 32;
        int row = rh >> 2, head = rh & 3, d0 = head * 16;
        int kb0 = row & 8;                         // key block for this pixel
        float logits[8];
        float mx = -3.4e38f;
#pragma unroll
        for (int j = 0; j < 8; ++j) {
            float s = 0.0f;
#pragma unroll
            for (int d = 0; d < 16; ++d) s += sQ[wid][row][d0 + d] * sK[wid][kb0 + j][d0 + d];
            s *= 0.25f;                            // hd^-0.5, hd=16
            if (!present[b * 8 + j]) s = -3.4e38f;
            logits[j] = s;
            if (s > mx) mx = s;
        }
        float den = 0.0f;
#pragma unroll
        for (int j = 0; j < 8; ++j) { logits[j] = __expf(logits[j] - mx); den += logits[j]; }
        float inv = 1.0f / den;
#pragma unroll
        for (int d = 0; d < 16; ++d) {
            float s = 0.0f;
#pragma unroll
            for (int j = 0; j < 8; ++j) s += logits[j] * sV[wid][kb0 + j][d0 + d];
            ao[it][d] = s * inv;
        }
    }
    __syncthreads();
#pragma unroll
    for (int it = 0; it < 2; ++it) {               // reuse sQ as AO staging
        int rh = lane + it * 32;
        int row = rh >> 2, head = rh & 3;
#pragma unroll
        for (int d = 0; d < 16; ++d) sQ[wid][row][head * 16 + d] = ao[it][d];
    }
    __syncthreads();

    // Build A-fragments for O-projection from sQ (f32 -> packed bf16).
    v8u t0, t1;
    {
        int row = lane & 15, koff = (lane & 16) ? 8 : 0;
#pragma unroll
        for (int d = 0; d < 8; ++d) {
            int g = d >> 2, vv = d & 3;
            int k0 = g * 16 + koff + 2 * vv;
            int k1 = 32 + k0;
            t0[d] = (unsigned int)f2bf(sQ[wid][row][k0]) |
                    ((unsigned int)f2bf(sQ[wid][row][k0 + 1]) << 16);
            t1[d] = (unsigned int)f2bf(sQ[wid][row][k1]) |
                    ((unsigned int)f2bf(sQ[wid][row][k1 + 1]) << 16);
        }
    }
    v16bf oa0 = __builtin_bit_cast(v16bf, t0);
    v16bf oa1 = __builtin_bit_cast(v16bf, t1);

    float v[4][8];
#pragma unroll
    for (int nt = 0; nt < 4; ++nt) {
        int n = nt * 16 + (lane & 15);
        float bv = ob[n];
        v8f c = {bv, bv, bv, bv, bv, bv, bv, bv};
        v16bf b0 = load_b_frag(Wo, nt, 0,  lane);
        v16bf b1 = load_b_frag(Wo, nt, 32, lane);
        c = __builtin_amdgcn_wmma_f32_16x16x32_bf16(false, oa0, false, b0, (short)0, c, false, false);
        c = __builtin_amdgcn_wmma_f32_16x16x32_bf16(false, oa1, false, b1, (short)0, c, false, false);
        int rb = r0 + ((lane & 16) ? 8 : 0);
#pragma unroll
        for (int i = 0; i < 8; ++i) v[nt][i] = c[i] + tok[(rb + i) * CH + n]; // residual
    }
    // LayerNorm per row (64 cols) via 16-lane half-wave reductions.
    int n15 = lane & 15;
#pragma unroll
    for (int i = 0; i < 8; ++i) {
        float s = v[0][i] + v[1][i] + v[2][i] + v[3][i];
        s += __shfl_xor(s, 1); s += __shfl_xor(s, 2);
        s += __shfl_xor(s, 4); s += __shfl_xor(s, 8);
        float mean = s * (1.0f / 64.0f);
        float q = 0.0f;
#pragma unroll
        for (int nt = 0; nt < 4; ++nt) { float d = v[nt][i] - mean; q += d * d; }
        q += __shfl_xor(q, 1); q += __shfl_xor(q, 2);
        q += __shfl_xor(q, 4); q += __shfl_xor(q, 8);
        float rstd = rsqrtf(q * (1.0f / 64.0f) + 1e-5f);
        int rowg = r0 + ((lane & 16) ? 8 : 0) + i;
#pragma unroll
        for (int nt = 0; nt < 4; ++nt) {
            int n = nt * 16 + n15;
            tok[rowg * CH + n] = (v[nt][i] - mean) * rstd * lng[n] + lnb[n];
        }
    }
}

// ---- layout transposes -----------------------------------------------------
__global__ void k_tok_from_conv(const float* __restrict__ x, float* __restrict__ tok,
                                unsigned short* __restrict__ Abf) {
    int idx = blockIdx.x * 256 + threadIdx.x;      // (img, c, hw) conv order
    int hw = idx & 1023, c = (idx >> 10) & 63, img = idx >> 16;
    int b = img >> 3, m = img & 7;
    float v = x[idx];
    int row = ((b << 10) + hw) * 8 + m;
    tok[row * CH + c] = v;
    Abf[row * CH + c] = f2bf(v);
}
__global__ void k_conv_from_tok(const float* __restrict__ tok, float* __restrict__ x) {
    int idx = blockIdx.x * 256 + threadIdx.x;
    int hw = idx & 1023, c = (idx >> 10) & 63, img = idx >> 16;
    int b = img >> 3, m = img & 7;
    x[idx] = tok[(((b << 10) + hw) * 8 + m) * CH + c];
}

// ---- depthwise 3x3, SAME padding ------------------------------------------
__global__ void k_dw(const float* __restrict__ x, const float* __restrict__ w,
                     float* __restrict__ out) {
    int idx = blockIdx.x * 256 + threadIdx.x;
    int hw = idx & 1023, c = (idx >> 10) & 63, img = idx >> 16;
    int h = hw >> 5, wc = hw & 31;
    const float* wp = w + c * 9;
    const float* xp = x + img * 65536 + c * 1024;
    float s = 0.0f;
#pragma unroll
    for (int ky = 0; ky < 3; ++ky) {
        int yy = h + ky - 1;
        if (yy < 0 || yy > 31) continue;
#pragma unroll
        for (int kx = 0; kx < 3; ++kx) {
            int xx = wc + kx - 1;
            if (xx < 0 || xx > 31) continue;
            s += xp[yy * 32 + xx] * wp[ky * 3 + kx];
        }
    }
    out[idx] = s;
}

// ---- group norm stats (8 groups of 8 channels, per image) ------------------
__global__ void k_gn_stats_conv(const float* __restrict__ x, float* __restrict__ stats) {
    int img = blockIdx.x >> 3, grp = blockIdx.x & 7;
    const float* p = x + img * 65536 + grp * 8192;
    float s = 0.0f, ss = 0.0f;
    for (int i = threadIdx.x; i < 8192; i += 256) { float v = p[i]; s += v; ss += v * v; }
    s  += __shfl_xor(s, 1);  s += __shfl_xor(s, 2);  s += __shfl_xor(s, 4);
    s  += __shfl_xor(s, 8);  s += __shfl_xor(s, 16);
    ss += __shfl_xor(ss, 1); ss += __shfl_xor(ss, 2); ss += __shfl_xor(ss, 4);
    ss += __shfl_xor(ss, 8); ss += __shfl_xor(ss, 16);
    __shared__ float rs[8], rss[8];
    if ((threadIdx.x & 31) == 0) { rs[threadIdx.x >> 5] = s; rss[threadIdx.x >> 5] = ss; }
    __syncthreads();
    if (threadIdx.x == 0) {
        float S = 0.0f, SS = 0.0f;
        for (int i = 0; i < 8; ++i) { S += rs[i]; SS += rss[i]; }
        float mean = S * (1.0f / 8192.0f);
        float var = SS * (1.0f / 8192.0f) - mean * mean;
        stats[blockIdx.x * 2] = mean;
        stats[blockIdx.x * 2 + 1] = rsqrtf(var + 1e-5f);
    }
}
__global__ void k_gn_stats_pm(const float* __restrict__ z, float* __restrict__ stats) {
    int img = blockIdx.x >> 3, grp = blockIdx.x & 7;
    int b = img >> 3, m = img & 7;
    float s = 0.0f, ss = 0.0f;
    for (int e = threadIdx.x; e < 8192; e += 256) {
        int n = e >> 3, j = e & 7;
        float v = z[(((b << 10) + n) * 8 + m) * CH + grp * 8 + j];
        s += v; ss += v * v;
    }
    s  += __shfl_xor(s, 1);  s += __shfl_xor(s, 2);  s += __shfl_xor(s, 4);
    s  += __shfl_xor(s, 8);  s += __shfl_xor(s, 16);
    ss += __shfl_xor(ss, 1); ss += __shfl_xor(ss, 2); ss += __shfl_xor(ss, 4);
    ss += __shfl_xor(ss, 8); ss += __shfl_xor(ss, 16);
    __shared__ float rs[8], rss[8];
    if ((threadIdx.x & 31) == 0) { rs[threadIdx.x >> 5] = s; rss[threadIdx.x >> 5] = ss; }
    __syncthreads();
    if (threadIdx.x == 0) {
        float S = 0.0f, SS = 0.0f;
        for (int i = 0; i < 8; ++i) { S += rs[i]; SS += rss[i]; }
        float mean = S * (1.0f / 8192.0f);
        float var = SS * (1.0f / 8192.0f) - mean * mean;
        stats[blockIdx.x * 2] = mean;
        stats[blockIdx.x * 2 + 1] = rsqrtf(var + 1e-5f);
    }
}

// stem: x_conv = silu(gn(tmp_pm)) (reads pixel-major, writes conv layout)
__global__ void k_stem_finish(const float* __restrict__ zpm, const float* __restrict__ stats,
                              const float* __restrict__ g, const float* __restrict__ bb,
                              float* __restrict__ x) {
    int idx = blockIdx.x * 256 + threadIdx.x;
    int hw = idx & 1023, c = (idx >> 10) & 63, img = idx >> 16;
    int b = img >> 3, m = img & 7;
    int sidx = (img * 8 + (c >> 3)) * 2;
    float z = zpm[(((b << 10) + hw) * 8 + m) * CH + c];
    float y = (z - stats[sidx]) * stats[sidx + 1] * g[c] + bb[c];
    x[idx] = siluf(y);
}
// gn1: Abf_pm = bf16(silu(gn(tmp_conv)))   (reads conv layout, writes pixel-major bf16)
__global__ void k_gn1_apply(const float* __restrict__ t, const float* __restrict__ stats,
                            const float* __restrict__ g, const float* __restrict__ bb,
                            unsigned short* __restrict__ Abf) {
    int idx = blockIdx.x * 256 + threadIdx.x;
    int hw = idx & 1023, c = (idx >> 10) & 63, img = idx >> 16;
    int b = img >> 3, m = img & 7;
    int sidx = (img * 8 + (c >> 3)) * 2;
    float y = (t[idx] - stats[sidx]) * stats[sidx + 1] * g[c] + bb[c];
    Abf[(((b << 10) + hw) * 8 + m) * CH + c] = f2bf(siluf(y));
}
// gn2 + residual + silu + present mask: x_conv = silu(x_conv + gn2(z_pm)) * present
__global__ void k_gn2_res(const float* __restrict__ zpm, const float* __restrict__ stats,
                          const float* __restrict__ g, const float* __restrict__ bb,
                          const int* __restrict__ present, float* __restrict__ x) {
    int idx = blockIdx.x * 256 + threadIdx.x;
    int hw = idx & 1023, c = (idx >> 10) & 63, img = idx >> 16;
    int b = img >> 3, m = img & 7;
    int sidx = (img * 8 + (c >> 3)) * 2;
    float z = zpm[(((b << 10) + hw) * 8 + m) * CH + c];
    float y = (z - stats[sidx]) * stats[sidx + 1] * g[c] + bb[c];
    float xc = siluf(x[idx] + y);
    x[idx] = present[img] ? xc : 0.0f;
}

// ---- heads -----------------------------------------------------------------
__global__ void k_policy(const float* __restrict__ x, const float* __restrict__ pw,
                         const float* __restrict__ pb, float* __restrict__ out) {
    int t = blockIdx.x * 256 + threadIdx.x;        // B*HW = 16384
    int b = t >> 10, hw = t & 1023;
    const float* xp = x + (b * 8) * 65536 + hw;    // player 0
    float s = pb[0];
#pragma unroll
    for (int c = 0; c < CH; ++c) s += xp[c * 1024] * pw[c];
    out[t] = s;
}
__global__ void k_pool(const float* __restrict__ x, float* __restrict__ pooled) {
    int img = blockIdx.x, c = threadIdx.x;         // 128 blocks x 64 threads
    const float* p = x + img * 65536 + c * 1024;
    float s = 0.0f;
    for (int i = 0; i < 1024; ++i) s += p[i];
    pooled[img * CH + c] = s * (1.0f / 1024.0f);
}
__global__ void k_value(const float* __restrict__ pooled, const float* __restrict__ w1,
                        const float* __restrict__ b1, const float* __restrict__ w2,
                        const float* __restrict__ b2, float* __restrict__ out) {
    int b = blockIdx.x, j = threadIdx.x;
    const float* p0 = pooled + (b * 8) * CH;
    float s = b1[j];
#pragma unroll
    for (int c = 0; c < CH; ++c) s += p0[c] * w1[j * CH + c];
    __shared__ float red[64];
    red[j] = siluf(s) * w2[j];
    __syncthreads();
    if (j == 0) {
        float S = b2[0];
        for (int k = 0; k < 64; ++k) S += red[k];
        out[b] = S;
    }
}
__global__ void k_opp_move(const float* __restrict__ x, const float* __restrict__ ow,
                           const float* __restrict__ ob, const int* __restrict__ present,
                           float* __restrict__ out) {
    int t = blockIdx.x * 256 + threadIdx.x;        // B*M*HW = 131072
    int b = t >> 13, i = (t >> 10) & 7, hw = t & 1023;
    const float* xp = x + (b * 8 + i) * 65536 + hw;
    float s = ob[i];
#pragma unroll
    for (int c = 0; c < CH; ++c) s += xp[c * 1024] * ow[i * CH + c];
    out[t] = present[b * 8 + i] ? s : 0.0f;
}
__global__ void k_opp_param(const float* __restrict__ pooled, const float* __restrict__ w1,
                            const float* __restrict__ b1, const float* __restrict__ w2,
                            const float* __restrict__ b2, const int* __restrict__ present,
                            float* __restrict__ out) {
    int img = blockIdx.x, j = threadIdx.x;         // 128 blocks x 64 threads
    int i = img & 7;
    const float* p = pooled + img * CH;
    float s = b1[j];
#pragma unroll
    for (int c = 0; c < CH; ++c) s += p[c] * w1[j * CH + c];
    __shared__ float h[64];
    h[j] = siluf(s);
    __syncthreads();
    if (j < 5) {
        int r = 5 * i + j;
        float S = b2[r];
#pragma unroll
        for (int c = 0; c < CH; ++c) S += h[c] * w2[r * CH + c];
        out[img * 5 + j] = present[img] ? S : 0.0f;
    }
}

// ---------------------------------------------------------------------------
extern "C" void kernel_launch(void* const* d_in, const int* in_sizes, int n_in,
                              void* d_out, int out_size, void* d_ws, size_t ws_size,
                              hipStream_t stream) {
    const float* board  = (const float*)d_in[0];
    const float* stem_w = (const float*)d_in[1];
    const float* stem_g = (const float*)d_in[2];
    const float* stem_b = (const float*)d_in[3];
    const float* qw     = (const float*)d_in[4];
    const float* qb     = (const float*)d_in[5];
    const float* kw     = (const float*)d_in[6];
    const float* kb     = (const float*)d_in[7];
    const float* vw     = (const float*)d_in[8];
    const float* vb     = (const float*)d_in[9];
    const float* ow     = (const float*)d_in[10];
    const float* ob     = (const float*)d_in[11];
    const float* ln_g   = (const float*)d_in[12];
    const float* ln_b   = (const float*)d_in[13];
    const float* dw_w   = (const float*)d_in[14];
    const float* gn1_g  = (const float*)d_in[15];
    const float* gn1_b  = (const float*)d_in[16];
    const float* pw_w   = (const float*)d_in[17];
    const float* gn2_g  = (const float*)d_in[18];
    const float* gn2_b  = (const float*)d_in[19];
    const float* pol_w  = (const float*)d_in[20];
    const float* pol_b  = (const float*)d_in[21];
    const float* vh_w1  = (const float*)d_in[22];
    const float* vh_b1  = (const float*)d_in[23];
    const float* vh_w2  = (const float*)d_in[24];
    const float* vh_b2  = (const float*)d_in[25];
    const float* om_w   = (const float*)d_in[26];
    const float* om_b   = (const float*)d_in[27];
    const float* op_w1  = (const float*)d_in[28];
    const float* op_b1  = (const float*)d_in[29];
    const float* op_w2  = (const float*)d_in[30];
    const float* op_b2  = (const float*)d_in[31];
    float* out = (float*)d_out;

    // workspace carve-up
    char* p = (char*)d_ws;
    float*          x_conv = (float*)p;          p += (size_t)RTOK * CH * 4;  // 32 MB
    float*          tok    = (float*)p;          p += (size_t)RTOK * CH * 4;  // 32 MB
    float*          tmp    = (float*)p;          p += (size_t)RTOK * CH * 4;  // 32 MB
    unsigned short* Abf    = (unsigned short*)p; p += (size_t)RTOK * CH * 2;  // 16 MB
    unsigned short* Wbf    = (unsigned short*)p; p += 262144;                 // 31*4096 bf16
    float*          stats  = (float*)p;          p += 8192;                   // 128*8*2
    int*            present= (int*)p;            p += 4096;                   // 128 ints
    float*          pooled = (float*)p;          p += 32768;                  // 128*64

    const int EW = RTOK * CH;        // 8388608 element-wise threads
    const int EWB = EW / 256;        // 32768 blocks

    k_present<<<1, 16, 0, stream>>>(board, present);
    k_prep_w<<<(31 * 4096 + 255) / 256, 256, 0, stream>>>(stem_w, qw, kw, vw, ow, pw_w, Wbf);

    // stem: gather -> WMMA GEMM -> GN(+silu) -> x_conv
    k_stem_gather<<<EWB, 256, 0, stream>>>(board, Abf);
    k_gemm<<<RTOK / 64, 128, 0, stream>>>(Abf, Wbf, nullptr, tmp, RTOK);
    k_gn_stats_pm<<<1024, 256, 0, stream>>>(tmp, stats);
    k_stem_finish<<<EWB, 256, 0, stream>>>(tmp, stats, stem_g, stem_b, x_conv);

    for (int l = 0; l < LAYERS; ++l) {
        const unsigned short* Wl = Wbf + (size_t)(1 + l * 5) * 4096;
        k_tok_from_conv<<<EWB, 256, 0, stream>>>(x_conv, tok, Abf);
        k_attn<<<RTOK / 64, 128, 0, stream>>>(Abf, tok,
            Wl, Wl + 4096, Wl + 2 * 4096, Wl + 3 * 4096,
            qb + l * 64, kb + l * 64, vb + l * 64, ob + l * 64,
            ln_g + l * 64, ln_b + l * 64, present);
        k_conv_from_tok<<<EWB, 256, 0, stream>>>(tok, x_conv);
        k_dw<<<EWB, 256, 0, stream>>>(x_conv, dw_w + l * 576, tmp);
        k_gn_stats_conv<<<1024, 256, 0, stream>>>(tmp, stats);
        k_gn1_apply<<<EWB, 256, 0, stream>>>(tmp, stats, gn1_g + l * 64, gn1_b + l * 64, Abf);
        k_gemm<<<RTOK / 64, 128, 0, stream>>>(Abf, Wl + 4 * 4096, nullptr, tok, RTOK);
        k_gn_stats_pm<<<1024, 256, 0, stream>>>(tok, stats);
        k_gn2_res<<<EWB, 256, 0, stream>>>(tok, stats, gn2_g + l * 64, gn2_b + l * 64, present, x_conv);
    }

    // heads: out = [logits_p 16384][value 16][opp_move 131072][opp_param 640]
    k_policy<<<16384 / 256, 256, 0, stream>>>(x_conv, pol_w, pol_b, out);
    k_pool<<<128, 64, 0, stream>>>(x_conv, pooled);
    k_value<<<16, 64, 0, stream>>>(pooled, vh_w1, vh_b1, vh_w2, vh_b2, out + 16384);
    k_opp_move<<<131072 / 256, 256, 0, stream>>>(x_conv, om_w, om_b, present, out + 16400);
    k_opp_param<<<128, 64, 0, stream>>>(pooled, op_w1, op_b1, op_w2, op_b2, present, out + 147472);
}
